// ModelOutput_73632919323329
// MI455X (gfx1250) — 8-bit matrix operand order repaired and checked by reading, compiled, not run on hardware
//
#include <hip/hip_runtime.h>
#include <hip/hip_bf16.h>
#include <math.h>

typedef int   v8i __attribute__((ext_vector_type(8)));
typedef float v4f __attribute__((ext_vector_type(4)));

#define K_TOTAL   65536
#define B_ROWS    2048
#define WQ_OFF    0
#define WMAX_OFF  65536
#define PART_OFF  65600

__global__ void wmax_kernel(const float* __restrict__ w, float* __restrict__ wmax) {
    __shared__ float red[256];
    float m = 0.0f;
    for (int i = threadIdx.x; i < K_TOTAL; i += 256)
        m = fmaxf(m, fabsf(w[i]));
    red[threadIdx.x] = m;
    __syncthreads();
    for (int s = 128; s > 0; s >>= 1) {
        if (threadIdx.x < s)
            red[threadIdx.x] = fmaxf(red[threadIdx.x], red[threadIdx.x + s]);
        __syncthreads();
    }
    if (threadIdx.x == 0) *wmax = red[0];
}

__global__ void wquant_kernel(const float* __restrict__ w,
                              const float* __restrict__ wmax,
                              char* __restrict__ wq) {
    const int i = (blockIdx.x * 256 + threadIdx.x) * 4;
    const float inv = 127.0f / fmaxf(*wmax, 1e-30f);
    char4 q;
    q.x = (char)(int)rintf(fminf(fmaxf(w[i + 0] * inv, -127.0f), 127.0f));
    q.y = (char)(int)rintf(fminf(fmaxf(w[i + 1] * inv, -127.0f), 127.0f));
    q.z = (char)(int)rintf(fminf(fmaxf(w[i + 2] * inv, -127.0f), 127.0f));
    q.w = (char)(int)rintf(fminf(fmaxf(w[i + 3] * inv, -127.0f), 127.0f));
    *(char4*)(wq + i) = q;
}

__global__ void zero_kernel(int* __restrict__ p) {
    p[blockIdx.x * 256 + threadIdx.x] = 0;
}

__device__ __forceinline__ int quant1(float x, float inv) {
    float t = x * inv;
    t = fminf(fmaxf(t, -127.0f), 127.0f);
    return (int)rintf(t);
}

__global__ void __launch_bounds__(256)
gemv_wmma_kernel(const float* __restrict__ x,
                 const char*  __restrict__ wq,
                 const float* __restrict__ clip,
                 int*         __restrict__ partial) {
    const int lane = threadIdx.x & 31;
    const int wave = threadIdx.x >> 5;
    const int m    = lane & 15;
    const int hi   = lane >> 4;

    const int rowBase = blockIdx.x * 16;
    const int split   = blockIdx.y * 8 + wave;
    const int kBegin  = split * (K_TOTAL / 64);

    const float c   = fabsf(clip[0]);
    const float inv = 127.0f / c;

    const float* xrow = x + (size_t)(rowBase + m) * (size_t)K_TOTAL;

    constexpr int abase[8] = {0, 4, 8, 12, 32, 36, 40, 44};

    v8i acc = {};

#pragma unroll 2
    for (int k0 = kBegin; k0 < kBegin + 1024; k0 += 64) {
        v8i a;
#pragma unroll
        for (int i = 0; i < 8; ++i) {
            const v4f f = __builtin_nontemporal_load(
                (const v4f*)(xrow + k0 + abase[i] + hi * 16));
            const int q0 = quant1(f.x, inv);
            const int q1 = quant1(f.y, inv);
            const int q2 = quant1(f.z, inv);
            const int q3 = quant1(f.w, inv);
            a[i] = (int)(((unsigned)(q0 & 0xff))       |
                         ((unsigned)(q1 & 0xff) << 8)  |
                         ((unsigned)(q2 & 0xff) << 16) |
                         ((unsigned)(q3 & 0xff) << 24));
        }
        const int4 blo = *(const int4*)(wq + k0 + hi * 16);
        const int4 bhi = *(const int4*)(wq + k0 + 32 + hi * 16);
        v8i b;
        b[0] = blo.x; b[1] = blo.y; b[2] = blo.z; b[3] = blo.w;
        b[4] = bhi.x; b[5] = bhi.y; b[6] = bhi.z; b[7] = bhi.w;

        acc = __builtin_amdgcn_wmma_i32_16x16x64_iu8(true, a, true, b, acc,
                                                     false, false);
    }

    if (m == 0) {
#pragma unroll
        for (int r = 0; r < 8; ++r)
            atomicAdd(&partial[rowBase + hi * 8 + r], acc[r]);
    }
}

__global__ void finalize_kernel(const int*   __restrict__ partial,
                                const float* __restrict__ bias,
                                const float* __restrict__ clip,
                                const float* __restrict__ wmax,
                                float*       __restrict__ out) {
    const int i = blockIdx.x * 256 + threadIdx.x;
    if (i < B_ROWS) {
        const float sx = fabsf(clip[0]) / 127.0f;
        const float sw = (*wmax) / 127.0f;
        out[i] = tanhf((float)partial[i] * sx * sw + bias[0]);
    }
}

extern "C" void kernel_launch(void* const* d_in, const int* in_sizes, int n_in,
                              void* d_out, int out_size, void* d_ws, size_t ws_size,
                              hipStream_t stream) {
    (void)in_sizes; (void)n_in; (void)out_size; (void)ws_size;
    const float* x    = (const float*)d_in[0];
    const float* w    = (const float*)d_in[1];
    const float* bias = (const float*)d_in[2];
    const float* clip = (const float*)d_in[3];
    float* out = (float*)d_out;

    char*  wq      = (char*)d_ws + WQ_OFF;
    float* wmax    = (float*)((char*)d_ws + WMAX_OFF);
    int*   partial = (int*)((char*)d_ws + PART_OFF);

    wmax_kernel  <<<1,   256, 0, stream>>>(w, wmax);
    wquant_kernel<<<64,  256, 0, stream>>>(w, wmax, wq);
    zero_kernel  <<<8,   256, 0, stream>>>(partial);
    gemv_wmma_kernel<<<dim3(B_ROWS / 16, 8), 256, 0, stream>>>(x, wq, clip, partial);
    finalize_kernel <<<8, 256, 0, stream>>>(partial, bias, clip, wmax, out);
}
